// VAD_Model_81192061763622
// MI455X (gfx1250) — matrix operand order repaired and checked by reading, not compiled, not run on hardware
//
#include <hip/hip_runtime.h>
#include <math.h>

// ---------------------------------------------------------------------------
// VAD model (2xLSTM + Mamba + memory heads) for MI455X / gfx1250, wave32.
// All big GEMMs use v_wmma_f32_16x16x32_bf16 with fp32 accumulation.
// Batch=16 maps exactly onto the WMMA M=16 tile, so the LSTM recurrence is a
// 1-M-tile GEMM per step with the 4 gates fused per wave.
// K-loops are branch-free and software-pipelined (double-buffered fragments)
// so vmem loads for tile k+1 overlap the WMMAs of tile k.
// Workspace usage: ~330 MB of d_ws (bf16 weight copies + sequences).
// ---------------------------------------------------------------------------

#define D_    2048
#define BSZ   16
#define T_    128
#define DS_   16
#define KCONV 4
#define MMEM  256
#define PJ_   128
#define G4D   (4 * D_)
#define BT    (BSZ * T_)          // 2048 rows

typedef __attribute__((ext_vector_type(16))) __bf16 v16bf;
typedef __attribute__((ext_vector_type(8)))  __bf16 v8bf;
typedef __attribute__((ext_vector_type(8)))  float  v8f;

static __device__ __forceinline__ float sigmoidf_(float x) {
  return 1.0f / (1.0f + __expf(-x));
}

static __device__ __forceinline__ v8f zero_v8f() {
  v8f z = {0.f, 0.f, 0.f, 0.f, 0.f, 0.f, 0.f, 0.f};
  return z;
}

// A-fragment (16x32 bf16, M x K). ISA layout: lane<16 holds row M=lane,
// K = {kb..kb+7, kb+16..kb+23} with kb = (lane>>4)*8.
static __device__ __forceinline__ v16bf
load_frag_a(const __bf16* __restrict__ A, int ld, int m0, int k0, int lane) {
  const __bf16* p = A + (size_t)(m0 + (lane & 15)) * ld + k0 + ((lane >> 4) << 3);
  v8bf lo = *(const v8bf*)(p);        // K offsets +0..7
  v8bf hi = *(const v8bf*)(p + 16);   // K offsets +16..23
  return __builtin_shufflevector(lo, hi, 0, 1, 2, 3, 4, 5, 6, 7,
                                 8, 9, 10, 11, 12, 13, 14, 15);
}

// B-fragment (32x16 bf16, K x N) sourced from row-major W[N,K] (i.e. B = W^T).
// ISA layout: lane holds column n = n0+(lane&15),
// K = {kb..kb+7, kb+16..kb+23} with kb = k0+(lane>>4)*8 (same as A).
static __device__ __forceinline__ v16bf
load_frag_b(const __bf16* __restrict__ W, int ld, int n0, int k0, int lane) {
  const __bf16* p = W + (size_t)(n0 + (lane & 15)) * ld + k0 + ((lane >> 4) << 3);
  v8bf lo = *(const v8bf*)(p);
  v8bf hi = *(const v8bf*)(p + 16);
  return __builtin_shufflevector(lo, hi, 0, 1, 2, 3, 4, 5, 6, 7,
                                 8, 9, 10, 11, 12, 13, 14, 15);
}

// ---------------------------------------------------------------------------
__global__ void f32_to_bf16_k(const float* __restrict__ in,
                              __bf16* __restrict__ out, int n) {
  int i = blockIdx.x * blockDim.x + threadIdx.x;
  if (i < n) out[i] = (__bf16)in[i];
}

// ---------------------------------------------------------------------------
// C[M,N] = A_bf16[M,K] @ W_bf16[N,K]^T + bias1[n] + bias2[n]
// One wave computes a 16x64 tile (4 N-subtiles sharing the A fragment).
// Tile bases are clamped (not branched); the K-loop is straight-line and
// double-buffered: loads for tile k+1 are issued before the WMMAs of tile k.
__global__ void gemm_bf16_nt(const __bf16* __restrict__ A,
                             const __bf16* __restrict__ W,
                             const float* __restrict__ bias1,
                             const float* __restrict__ bias2,
                             float* __restrict__ C,
                             int M, int N, int K, int ldc) {
  const int lane = threadIdx.x & 31;
  const int wid  = __builtin_amdgcn_readfirstlane(
      blockIdx.x * (blockDim.x >> 5) + (threadIdx.x >> 5));
  const int n64  = (N + 63) >> 6;
  const int mT   = M >> 4;
  if (wid >= mT * n64) return;
  const int mt = wid / n64;
  const int nt = wid % n64;
  const int m0 = mt << 4;

  // Clamped (always in-bounds) N-subtile bases; duplicates are discarded in
  // the guarded epilogue, keeping the hot loop branch-free.
  int n0c[4];
#pragma unroll
  for (int j = 0; j < 4; ++j) {
    int v = (nt << 6) + (j << 4);
    n0c[j] = (v < N) ? v : (N - 16);
  }

  v8f acc[4];
#pragma unroll
  for (int j = 0; j < 4; ++j) acc[j] = zero_v8f();

  const int KT = K >> 5;                      // K-tiles of 32
  v16bf a_cur = load_frag_a(A, K, m0, 0, lane);
  v16bf b_cur[4];
#pragma unroll
  for (int j = 0; j < 4; ++j) b_cur[j] = load_frag_b(W, K, n0c[j], 0, lane);

  for (int kt = 0; kt + 1 < KT; ++kt) {
    const int k1 = (kt + 1) << 5;
    // prefetch next tile into fresh registers (overlaps the WMMAs below)
    v16bf a_nxt = load_frag_a(A, K, m0, k1, lane);
    v16bf b_nxt[4];
#pragma unroll
    for (int j = 0; j < 4; ++j) b_nxt[j] = load_frag_b(W, K, n0c[j], k1, lane);
#pragma unroll
    for (int j = 0; j < 4; ++j)
      acc[j] = __builtin_amdgcn_wmma_f32_16x16x32_bf16(
          false, a_cur, false, b_cur[j], (short)0, acc[j], false, false);
    a_cur = a_nxt;
#pragma unroll
    for (int j = 0; j < 4; ++j) b_cur[j] = b_nxt[j];
  }
#pragma unroll
  for (int j = 0; j < 4; ++j)
    acc[j] = __builtin_amdgcn_wmma_f32_16x16x32_bf16(
        false, a_cur, false, b_cur[j], (short)0, acc[j], false, false);

  const int mrow = m0 + ((lane >> 4) << 3);
  const int ncol = lane & 15;
#pragma unroll
  for (int j = 0; j < 4; ++j) {
    int n0 = (nt << 6) + (j << 4);
    if (n0 >= N) continue;                 // scalar (nt is SGPR)
    int n = n0 + ncol;
    float bv = (bias1 ? bias1[n] : 0.f) + (bias2 ? bias2[n] : 0.f);
#pragma unroll
    for (int r = 0; r < 8; ++r)
      C[(size_t)(mrow + r) * ldc + n] = acc[j][r] + bv;
  }
}

// ---------------------------------------------------------------------------
// One LSTM timestep: g = xg[:,t,:] + h_{t-1} @ Whh^T, then the cell update.
// 128 waves; wave w owns hidden columns j0 = 16*w and all 4 gates (fused).
// A fragment = h_{t-1}[16 x 32] (batch=16 rows == WMMA M tile), bf16 from the
// sequence buffer written at step t-1 (row stride T*D). For t==0 (h0 = 0) the
// whole GEMM loop is skipped and acc stays zero.
__global__ void lstm_step_k(const float* __restrict__ xg,      // [BT, 4D]
                            const __bf16* __restrict__ whh,    // [4D, D] bf16
                            __bf16* __restrict__ hseq_bf,      // [BT, D]
                            float* __restrict__ hseq_f32,      // [BT, D] or null
                            float* __restrict__ c_state,       // [B, D]
                            int t) {
  const int lane = threadIdx.x & 31;
  const int wid  = __builtin_amdgcn_readfirstlane(
      blockIdx.x * (blockDim.x >> 5) + (threadIdx.x >> 5));
  const int j0   = wid << 4;

  v8f acc[4];
#pragma unroll
  for (int g = 0; g < 4; ++g) acc[g] = zero_v8f();

  if (t > 0) {                              // scalar branch (t is uniform)
    const __bf16* hprev = hseq_bf + (size_t)(t - 1) * D_;  // row stride T_*D_
    const int KT = D_ >> 5;
    v16bf a_cur = load_frag_a(hprev, T_ * D_, 0, 0, lane);
    v16bf b_cur[4];
#pragma unroll
    for (int g = 0; g < 4; ++g)
      b_cur[g] = load_frag_b(whh, D_, g * D_ + j0, 0, lane);

    for (int kt = 0; kt + 1 < KT; ++kt) {
      const int k1 = (kt + 1) << 5;
      v16bf a_nxt = load_frag_a(hprev, T_ * D_, 0, k1, lane);
      v16bf b_nxt[4];
#pragma unroll
      for (int g = 0; g < 4; ++g)
        b_nxt[g] = load_frag_b(whh, D_, g * D_ + j0, k1, lane);
#pragma unroll
      for (int g = 0; g < 4; ++g)
        acc[g] = __builtin_amdgcn_wmma_f32_16x16x32_bf16(
            false, a_cur, false, b_cur[g], (short)0, acc[g], false, false);
      a_cur = a_nxt;
#pragma unroll
      for (int g = 0; g < 4; ++g) b_cur[g] = b_nxt[g];
    }
#pragma unroll
    for (int g = 0; g < 4; ++g)
      acc[g] = __builtin_amdgcn_wmma_f32_16x16x32_bf16(
          false, a_cur, false, b_cur[g], (short)0, acc[g], false, false);
  }

  const int hi = lane >> 4;
  const int n  = j0 + (lane & 15);
#pragma unroll
  for (int r = 0; r < 8; ++r) {
    int b   = hi * 8 + r;            // batch index (WMMA C-layout M index)
    int row = b * T_ + t;
    const float* xgp = xg + (size_t)row * G4D;
    float gi = acc[0][r] + xgp[n];
    float gf = acc[1][r] + xgp[D_ + n];
    float gg = acc[2][r] + xgp[2 * D_ + n];
    float go = acc[3][r] + xgp[3 * D_ + n];
    float cp = (t == 0) ? 0.f : c_state[b * D_ + n];
    float c  = sigmoidf_(gf) * cp + sigmoidf_(gi) * tanhf(gg);
    float h  = sigmoidf_(go) * tanhf(c);
    c_state[b * D_ + n] = c;
    hseq_bf[(size_t)row * D_ + n] = (__bf16)h;
    if (hseq_f32) hseq_f32[(size_t)row * D_ + n] = h;
  }
}

// ---------------------------------------------------------------------------
// Causal depthwise conv (K=4, left pad 3) over xm = xz[:, :D], then SiLU.
__global__ void conv_silu_k(const float* __restrict__ xz,   // [BT, 2D]
                            const float* __restrict__ cw,   // [D, 1, K]
                            const float* __restrict__ cb,   // [D]
                            float* __restrict__ xm,         // [BT, D]
                            __bf16* __restrict__ xm_bf) {
  int idx = blockIdx.x * blockDim.x + threadIdx.x;
  if (idx >= BT * D_) return;
  int d = idx & (D_ - 1);
  int row = idx >> 11;              // / D_
  int t = row & (T_ - 1);
  int b = row >> 7;                 // / T_
  float s = cb[d];
#pragma unroll
  for (int k = 0; k < KCONV; ++k) {
    int tt = t + k - (KCONV - 1);
    if (tt >= 0)
      s += cw[d * KCONV + k] * xz[(size_t)(b * T_ + tt) * (2 * D_) + d];
  }
  float v = s * sigmoidf_(s);       // SiLU
  xm[idx] = v;
  xm_bf[idx] = (__bf16)v;
}

// ---------------------------------------------------------------------------
// Tiny SSM scan: h_t = h_{t-1} @ A^T + u_t * B_ssm ; s_t[b] = h_t[b,:] . C_ssm
// One block of 256 threads = (batch 16) x (state 16).
__global__ void ssm_scan_k(const float* __restrict__ u,     // [BT, DS]
                           const float* __restrict__ Amat,  // [DS, DS]
                           const float* __restrict__ Bssm,  // [DS]
                           const float* __restrict__ Cssm,  // [DS]
                           float* __restrict__ s_out) {     // [B*T] as [b,t]
  __shared__ float hbuf[BSZ][DS_ + 1];
  __shared__ float sc[BSZ][DS_ + 1];
  int tid = threadIdx.x;
  int b = tid >> 4, ds = tid & 15;
  float arow[DS_];
#pragma unroll
  for (int k = 0; k < DS_; ++k) arow[k] = Amat[ds * DS_ + k];
  float bscale = Bssm[ds], cval = Cssm[ds];
  hbuf[b][ds] = 0.f;
  __syncthreads();
  for (int t = 0; t < T_; ++t) {
    float hv = u[(size_t)(b * T_ + t) * DS_ + ds] * bscale;
#pragma unroll
    for (int k = 0; k < DS_; ++k) hv += hbuf[b][k] * arow[k];
    __syncthreads();
    hbuf[b][ds] = hv;
    sc[b][ds] = hv * cval;
    __syncthreads();
    if (ds == 0) {
      float acc = 0.f;
#pragma unroll
      for (int k = 0; k < DS_; ++k) acc += sc[b][k];
      s_out[b * T_ + t] = acc;
    }
  }
}

// ---------------------------------------------------------------------------
// y_last[b,d] = s[b,T-1] * xm[b,T-1,d] * silu(gate[b,T-1,d])
__global__ void ylast_k(const float* __restrict__ xz, const float* __restrict__ xm,
                        const float* __restrict__ s,
                        float* __restrict__ ylast, __bf16* __restrict__ ylast_bf) {
  int idx = blockIdx.x * blockDim.x + threadIdx.x;
  if (idx >= BSZ * D_) return;
  int b = idx >> 11, d = idx & (D_ - 1);
  int row = b * T_ + (T_ - 1);
  float g = xz[(size_t)row * (2 * D_) + D_ + d];
  float y = s[b * T_ + (T_ - 1)] * xm[(size_t)row * D_ + d] * (g * sigmoidf_(g));
  ylast[idx] = y;
  ylast_bf[idx] = (__bf16)y;
}

// feat = lstm_out[:, T-1, :] + 0.2 * mamba_last
__global__ void feat_k(const float* __restrict__ lstm_f32,
                       const float* __restrict__ mamba_last,
                       float* __restrict__ feat, __bf16* __restrict__ feat_bf) {
  int idx = blockIdx.x * blockDim.x + threadIdx.x;
  if (idx >= BSZ * D_) return;
  int b = idx >> 11, d = idx & (D_ - 1);
  int row = b * T_ + (T_ - 1);
  float v = lstm_f32[(size_t)row * D_ + d] + 0.2f * mamba_last[idx];
  feat[idx] = v;
  feat_bf[idx] = (__bf16)v;
}

// anomaly (min L2 dist over prim+sec memories) and fc head. One block/batch.
__global__ void heads_k(const float* __restrict__ feat,
                        const float* __restrict__ prim, const float* __restrict__ sec,
                        const float* __restrict__ fcw, const float* __restrict__ fcb,
                        float* __restrict__ out) {
  __shared__ float red[256];
  int b = blockIdx.x, tid = threadIdx.x;
  const float* f = feat + (size_t)b * D_;
  float best = 3.4e38f;
  for (int m = 0; m < 2 * MMEM; ++m) {
    const float* mem = (m < MMEM) ? prim + (size_t)m * D_
                                  : sec + (size_t)(m - MMEM) * D_;
    float p = 0.f;
    for (int d = tid; d < D_; d += 256) {
      float df = mem[d] - f[d];
      p += df * df;
    }
    red[tid] = p;
    __syncthreads();
    for (int s = 128; s > 0; s >>= 1) {
      if (tid < s) red[tid] += red[tid + s];
      __syncthreads();
    }
    if (tid == 0) best = fminf(best, sqrtf(red[0]));
    __syncthreads();
  }
  float p = 0.f;
  for (int d = tid; d < D_; d += 256) p += f[d] * fcw[d];
  red[tid] = p;
  __syncthreads();
  for (int s = 128; s > 0; s >>= 1) {
    if (tid < s) red[tid] += red[tid + s];
    __syncthreads();
  }
  if (tid == 0) {
    out[b] = red[0] + fcb[0];       // fc output [16,1]
    out[BSZ + b] = best;            // anomaly [16]
  }
}

// ---------------------------------------------------------------------------
extern "C" void kernel_launch(void* const* d_in, const int* in_sizes, int n_in,
                              void* d_out, int out_size, void* d_ws, size_t ws_size,
                              hipStream_t stream) {
  (void)in_sizes; (void)n_in; (void)out_size; (void)ws_size;

  const float* x         = (const float*)d_in[0];
  const float* w_ih0     = (const float*)d_in[1];
  const float* w_hh0     = (const float*)d_in[2];
  const float* b_ih0     = (const float*)d_in[3];
  const float* b_hh0     = (const float*)d_in[4];
  const float* w_ih1     = (const float*)d_in[5];
  const float* w_hh1     = (const float*)d_in[6];
  const float* b_ih1     = (const float*)d_in[7];
  const float* b_hh1     = (const float*)d_in[8];
  const float* in_proj_w = (const float*)d_in[9];
  const float* in_proj_b = (const float*)d_in[10];
  const float* conv_w    = (const float*)d_in[11];
  const float* conv_b    = (const float*)d_in[12];
  const float* ssm_w     = (const float*)d_in[13];
  const float* ssm_b     = (const float*)d_in[14];
  const float* A_mat     = (const float*)d_in[15];
  const float* B_ssm     = (const float*)d_in[16];
  const float* C_ssm     = (const float*)d_in[17];
  const float* out_proj_w = (const float*)d_in[18];
  const float* out_proj_b = (const float*)d_in[19];
  const float* prim_mem  = (const float*)d_in[20];
  const float* sec_mem   = (const float*)d_in[21];
  const float* fc_w      = (const float*)d_in[22];
  const float* fc_b      = (const float*)d_in[23];
  const float* proj_w    = (const float*)d_in[24];
  const float* proj_b    = (const float*)d_in[25];
  float* out = (float*)d_out;

  // ---- linear workspace allocation (deterministic, rewritten every call) ---
  char* ws = (char*)d_ws;
  size_t off = 0;
  auto alloc = [&](size_t bytes) -> char* {
    char* p = ws + off;
    off += (bytes + 255) & ~(size_t)255;
    return p;
  };
  __bf16* xbf        = (__bf16*)alloc((size_t)BT * D_ * 2);
  __bf16* wih0bf     = (__bf16*)alloc((size_t)G4D * D_ * 2);
  __bf16* whh0bf     = (__bf16*)alloc((size_t)G4D * D_ * 2);
  __bf16* wih1bf     = (__bf16*)alloc((size_t)G4D * D_ * 2);
  __bf16* whh1bf     = (__bf16*)alloc((size_t)G4D * D_ * 2);
  __bf16* inprojbf   = (__bf16*)alloc((size_t)2 * D_ * D_ * 2);
  __bf16* ssmbf      = (__bf16*)alloc((size_t)DS_ * D_ * 2);
  __bf16* outprojbf  = (__bf16*)alloc((size_t)D_ * D_ * 2);
  __bf16* projwbf    = (__bf16*)alloc((size_t)PJ_ * D_ * 2);
  __bf16* h1seq_bf   = (__bf16*)alloc((size_t)BT * D_ * 2);
  __bf16* loutseq_bf = (__bf16*)alloc((size_t)BT * D_ * 2);
  __bf16* xm_bf      = (__bf16*)alloc((size_t)BT * D_ * 2);
  __bf16* ylast_bf   = (__bf16*)alloc((size_t)BSZ * D_ * 2);
  __bf16* feat_bf    = (__bf16*)alloc((size_t)BSZ * D_ * 2);
  float* gates       = (float*)alloc((size_t)BT * G4D * 4);
  float* loutseq_f   = (float*)alloc((size_t)BT * D_ * 4);
  float* c_state     = (float*)alloc((size_t)BSZ * D_ * 4);
  float* xz          = (float*)alloc((size_t)BT * 2 * D_ * 4);
  float* xm          = (float*)alloc((size_t)BT * D_ * 4);
  float* u_buf       = (float*)alloc((size_t)BT * DS_ * 4);
  float* s_buf       = (float*)alloc((size_t)BT * 4);
  float* ylast       = (float*)alloc((size_t)BSZ * D_ * 4);
  float* mamba_last  = (float*)alloc((size_t)BSZ * D_ * 4);
  float* feat        = (float*)alloc((size_t)BSZ * D_ * 4);

  auto cvt = [&](const float* src, __bf16* dst, int n) {
    f32_to_bf16_k<<<(n + 255) / 256, 256, 0, stream>>>(src, dst, n);
  };
  auto gemm = [&](const __bf16* A, const __bf16* W, const float* b1,
                  const float* b2, float* C, int M, int N, int K, int ldc) {
    int waves = (M >> 4) * ((N + 63) >> 6);
    int blocks = (waves + 7) / 8;
    gemm_bf16_nt<<<blocks, 256, 0, stream>>>(A, W, b1, b2, C, M, N, K, ldc);
  };

  // 1) bf16 conversions (amortized: each weight re-used by 128 tiles/steps)
  cvt(x, xbf, BT * D_);
  cvt(w_ih0, wih0bf, G4D * D_);
  cvt(w_hh0, whh0bf, G4D * D_);
  cvt(w_ih1, wih1bf, G4D * D_);
  cvt(w_hh1, whh1bf, G4D * D_);
  cvt(in_proj_w, inprojbf, 2 * D_ * D_);
  cvt(ssm_w, ssmbf, DS_ * D_);
  cvt(out_proj_w, outprojbf, D_ * D_);
  cvt(proj_w, projwbf, PJ_ * D_);

  // 2) LSTM layer 0: xg = x @ w_ih0^T + b_ih0 + b_hh0, then 128 fused steps
  gemm(xbf, wih0bf, b_ih0, b_hh0, gates, BT, G4D, D_, G4D);
  for (int t = 0; t < T_; ++t)
    lstm_step_k<<<16, 256, 0, stream>>>(gates, whh0bf, h1seq_bf, nullptr,
                                        c_state, t);

  // 3) LSTM layer 1 (c_state re-seeded via t==0 path; no reset needed)
  gemm(h1seq_bf, wih1bf, b_ih1, b_hh1, gates, BT, G4D, D_, G4D);
  for (int t = 0; t < T_; ++t)
    lstm_step_k<<<16, 256, 0, stream>>>(gates, whh1bf, loutseq_bf, loutseq_f,
                                        c_state, t);

  // 4) Mamba
  gemm(loutseq_bf, inprojbf, in_proj_b, nullptr, xz, BT, 2 * D_, D_, 2 * D_);
  conv_silu_k<<<(BT * D_ + 255) / 256, 256, 0, stream>>>(xz, conv_w, conv_b,
                                                         xm, xm_bf);
  gemm(xm_bf, ssmbf, ssm_b, nullptr, u_buf, BT, DS_, D_, DS_);
  ssm_scan_k<<<1, 256, 0, stream>>>(u_buf, A_mat, B_ssm, C_ssm, s_buf);
  ylast_k<<<(BSZ * D_ + 255) / 256, 256, 0, stream>>>(xz, xm, s_buf, ylast,
                                                      ylast_bf);
  gemm(ylast_bf, outprojbf, out_proj_b, nullptr, mamba_last, BSZ, D_, D_, D_);

  // 5) feat + heads
  feat_k<<<(BSZ * D_ + 255) / 256, 256, 0, stream>>>(loutseq_f, mamba_last,
                                                     feat, feat_bf);
  gemm(feat_bf, projwbf, proj_b, nullptr, out + 2 * BSZ, BSZ, PJ_, D_, PJ_);
  heads_k<<<BSZ, 256, 0, stream>>>(feat, prim_mem, sec_mem, fc_w, fc_b, out);
}
